// PointNetSetAbstraction_86921548136941
// MI455X (gfx1250) — matrix operand order repaired and checked by reading, not compiled, not run on hardware
//
#include <hip/hip_runtime.h>

// ---------------------------------------------------------------------------
// PointNet Set Abstraction for MI455X (gfx1250, wave32, WMMA)
// B=16, N=8192, G=512, K=32, D=128, MLP=[128,128,256]
// ---------------------------------------------------------------------------

#define Bc   16
#define Nc   8192
#define Gc   512
#define Kc   32
#define Dc   128
#define MTOT (Bc * Kc * Gc)          // 262144 spatial elements (b,k,g)
#define FINF 3.0e38f

typedef __attribute__((ext_vector_type(16))) _Float16 v16h;
typedef __attribute__((ext_vector_type(8)))  float    v8f;

union AFrag { v16h h; uint32_t u[8]; };
union H8   { uint4 v; _Float16 h[8]; };

// ---------------------------------------------------------------------------
// 1) Farthest point sampling: one block per batch, dists in LDS,
//    511 sequential block-wide argmax reductions. Writes centers [B][G][3].
// ---------------------------------------------------------------------------
__global__ __launch_bounds__(1024) void fps_kernel(const float* __restrict__ xyz,
                                                   float* __restrict__ centers) {
    const int b = blockIdx.x;
    const float* base = xyz + (size_t)b * Nc * 3;
    __shared__ float dist[Nc];
    __shared__ float redv[32];
    __shared__ int   redi[32];
    __shared__ float scx, scy, scz;

    const int tid  = threadIdx.x;
    const int lane = tid & 31;
    const int wid  = tid >> 5;

    for (int i = tid; i < Nc; i += 1024) dist[i] = 1e10f;
    if (tid == 0) {
        scx = base[0]; scy = base[1]; scz = base[2];
        size_t o = (size_t)b * Gc * 3;
        centers[o + 0] = scx; centers[o + 1] = scy; centers[o + 2] = scz;
    }
    __syncthreads();

    for (int it = 1; it < Gc; ++it) {
        const float cx = scx, cy = scy, cz = scz;
        float bv = -1.0f; int bi = 0x7fffffff;
        for (int i = tid; i < Nc; i += 1024) {
            float dx = base[i * 3 + 0] - cx;
            float dy = base[i * 3 + 1] - cy;
            float dz = base[i * 3 + 2] - cz;
            float d  = dx * dx + dy * dy + dz * dz;
            float nd = fminf(dist[i], d);
            dist[i] = nd;
            if (nd > bv) { bv = nd; bi = i; }
        }
        #pragma unroll
        for (int off = 16; off > 0; off >>= 1) {
            float ov = __shfl_down(bv, off, 32);
            int   oi = __shfl_down(bi, off, 32);
            if (ov > bv || (ov == bv && oi < bi)) { bv = ov; bi = oi; }
        }
        if (lane == 0) { redv[wid] = bv; redi[wid] = bi; }
        __syncthreads();
        if (wid == 0) {
            bv = redv[lane]; bi = redi[lane];
            #pragma unroll
            for (int off = 16; off > 0; off >>= 1) {
                float ov = __shfl_down(bv, off, 32);
                int   oi = __shfl_down(bi, off, 32);
                if (ov > bv || (ov == bv && oi < bi)) { bv = ov; bi = oi; }
            }
            if (lane == 0) {
                float nx = base[bi * 3 + 0];
                float ny = base[bi * 3 + 1];
                float nz = base[bi * 3 + 2];
                scx = nx; scy = ny; scz = nz;
                size_t o = ((size_t)b * Gc + it) * 3;
                centers[o + 0] = nx; centers[o + 1] = ny; centers[o + 2] = nz;
            }
        }
        __syncthreads();
    }
}

// ---------------------------------------------------------------------------
// 2) KNN top-32 smallest per (b,g): distances cached in LDS; 32 rounds of
//    incremental argmin (only the winning thread rescans its chunk).
// ---------------------------------------------------------------------------
__global__ __launch_bounds__(256) void knn_kernel(const float* __restrict__ xyz,
                                                  const float* __restrict__ centers,
                                                  int* __restrict__ knn_idx) {
    const int bg = blockIdx.x;             // b*G + g
    const int b  = bg >> 9;                // G = 512
    const float* base = xyz + (size_t)b * Nc * 3;
    __shared__ float d[Nc];
    __shared__ float redv[8];
    __shared__ int   redi[8];
    __shared__ int   swi;

    const int tid  = threadIdx.x;
    const int lane = tid & 31;
    const int wid  = tid >> 5;

    const float cx = centers[(size_t)bg * 3 + 0];
    const float cy = centers[(size_t)bg * 3 + 1];
    const float cz = centers[(size_t)bg * 3 + 2];

    float mv = FINF; int mi = Nc;
    for (int i = tid; i < Nc; i += 256) {
        float dx = base[i * 3 + 0] - cx;
        float dy = base[i * 3 + 1] - cy;
        float dz = base[i * 3 + 2] - cz;
        float dd = dx * dx + dy * dy + dz * dz;
        d[i] = dd;
        if (dd < mv) { mv = dd; mi = i; }
    }
    __syncthreads();

    for (int kk = 0; kk < Kc; ++kk) {
        float v = mv; int ix = mi;
        #pragma unroll
        for (int off = 16; off > 0; off >>= 1) {
            float ov = __shfl_down(v, off, 32);
            int   oi = __shfl_down(ix, off, 32);
            if (ov < v || (ov == v && oi < ix)) { v = ov; ix = oi; }
        }
        if (lane == 0) { redv[wid] = v; redi[wid] = ix; }
        __syncthreads();
        if (tid == 0) {
            float bv2 = redv[0]; int bi2 = redi[0];
            #pragma unroll
            for (int w = 1; w < 8; ++w)
                if (redv[w] < bv2 || (redv[w] == bv2 && redi[w] < bi2)) { bv2 = redv[w]; bi2 = redi[w]; }
            swi = bi2;
            knn_idx[(size_t)bg * Kc + kk] = bi2;
        }
        __syncthreads();
        const int w = swi;
        if ((w & 255) == tid) {            // winner owns this strided slot
            d[w] = FINF;
            mv = FINF; mi = Nc;
            for (int i = tid; i < Nc; i += 256) {
                float dd = d[i];
                if (dd < mv) { mv = dd; mi = i; }
            }
        }
        __syncthreads();
    }
}

// ---------------------------------------------------------------------------
// 3) Gather neighbor features to X0 f16 [m][c], m = (b*K + k)*G + g.
// ---------------------------------------------------------------------------
__global__ __launch_bounds__(256) void gather_kernel(const float* __restrict__ points,
                                                     const int* __restrict__ knn_idx,
                                                     _Float16* __restrict__ X0) {
    const int rid = blockIdx.x * 2 + (threadIdx.x >> 7);
    const int c   = threadIdx.x & 127;
    const int g   = rid & (Gc - 1);
    const int bk  = rid >> 9;
    const int k   = bk & (Kc - 1);
    const int b   = bk >> 5;
    const int src = knn_idx[((size_t)(b * Gc + g)) * Kc + k];
    float v = points[((size_t)b * Nc + src) * Dc + c];
    X0[(size_t)rid * Dc + c] = (_Float16)v;
}

// ---------------------------------------------------------------------------
// 4) f32 -> f16 weight conversion.
// ---------------------------------------------------------------------------
__global__ void cvt_f16_kernel(const float* __restrict__ src, _Float16* __restrict__ dst, int n) {
    int i = blockIdx.x * 256 + threadIdx.x;
    if (i < n) dst[i] = (_Float16)src[i];
}

// ---------------------------------------------------------------------------
// 5) WMMA GEMM: Y[m][oc] = (fused norm/relu on Xin)[m][c] * W[oc][c]^T + bias.
//    Block = 8 waves; each wave owns 16 output channels. Four 16-wide m-tiles
//    per group (4 independent accumulator chains to hide WMMA latency), all
//    B-fragments of a K-chunk loaded into distinct VGPRs so LDS loads overlap
//    matrix ops, and the next group's global tile is register-prefetched.
// ---------------------------------------------------------------------------
#define GEMM_MTILES 8     // m-tiles per block (processed in groups of 4)
#define BT_STRIDE   136   // 128 halves + pad

__global__ __launch_bounds__(256) void gemm_kernel(
    const _Float16* __restrict__ Xin,   // [MTOT][128] f16 (raw, pre-norm)
    const _Float16* __restrict__ W16,   // [OC][128]   f16
    const float*    __restrict__ bias,  // [OC]
    const float*    __restrict__ nscale,// [128] per-input-channel BN scale
    const float*    __restrict__ nshift,// [128] per-input-channel BN shift
    int apply_norm, int OC,
    _Float16* __restrict__ Yout)        // [MTOT][OC] f16
{
    __shared__ _Float16 bt[64 * BT_STRIDE];   // four 16-row m-tiles
    __shared__ _Float16 st[8][16 * 16];

    const int tid    = threadIdx.x;
    const int wave   = tid >> 5;
    const int lane   = tid & 31;
    const int hi     = lane >> 4;
    const int l15    = lane & 15;
    const int ocbase = blockIdx.y * 128 + wave * 16;

    // --- Preload A fragments (weights) for the 4 K-chunks, ISA 16b A layout:
    //     dword j holds halves K, K+1 with K = 32*cc + 2j + (j>=4)*8 + hi*8.
    AFrag a[4];
    {
        const _Float16* wr = W16 + (size_t)(ocbase + l15) * 128;
        #pragma unroll
        for (int cc = 0; cc < 4; ++cc) {
            #pragma unroll
            for (int j = 0; j < 8; ++j) {
                int k = cc * 32 + 2 * j + ((j >= 4) ? 8 : 0) + hi * 8;
                a[cc].u[j] = *(const uint32_t*)(wr + k);
            }
        }
    }
    // Bias per accumulator row (VGPR r -> oc = ocbase + r + hi*8).
    float bv[8];
    #pragma unroll
    for (int r = 0; r < 8; ++r) bv[r] = bias[ocbase + r + hi * 8];

    // B-tile cooperative-load coordinates: 256 threads x 4 uint4 cover
    // 64 rows x 128 channels. Thread handles rows r0 + 16h, same cols.
    const int r0 = tid >> 4;
    const int c0 = (tid & 15) * 8;
    float sc[8], sh[8];
    #pragma unroll
    for (int j = 0; j < 8; ++j) { sc[j] = 1.0f; sh[j] = 0.0f; }
    if (apply_norm) {
        #pragma unroll
        for (int j = 0; j < 8; ++j) { sc[j] = nscale[c0 + j]; sh[j] = nshift[c0 + j]; }
    }

    const int ngroups = GEMM_MTILES / 4;       // 2
    const int mbase   = blockIdx.x * GEMM_MTILES * 16;

    // Prefetch group 0's global tile into registers.
    H8 raw[4];
    #pragma unroll
    for (int h = 0; h < 4; ++h)
        raw[h].v = *(const uint4*)(Xin + (size_t)(mbase + r0 + 16 * h) * 128 + c0);

    for (int t = 0; t < ngroups; ++t) {
        const int mt = mbase + t * 64;

        // Stage the 4 m-tiles (apply previous layer's BN + ReLU on the fly).
        #pragma unroll
        for (int h = 0; h < 4; ++h) {
            H8 w = raw[h];
            if (apply_norm) {
                #pragma unroll
                for (int j = 0; j < 8; ++j) {
                    float v = (float)w.h[j] * sc[j] + sh[j];
                    w.h[j] = (_Float16)(v > 0.0f ? v : 0.0f);
                }
            }
            *(uint4*)&bt[(r0 + 16 * h) * BT_STRIDE + c0] = w.v;
        }
        __syncthreads();

        // Register-prefetch the next group's tile during compute.
        if (t + 1 < ngroups) {
            const int mtn = mbase + (t + 1) * 64;
            #pragma unroll
            for (int h = 0; h < 4; ++h)
                raw[h].v = *(const uint4*)(Xin + (size_t)(mtn + r0 + 16 * h) * 128 + c0);
        }

        v8f zero = {};
        v8f acc[4] = {zero, zero, zero, zero};
        #pragma unroll
        for (int cc = 0; cc < 4; ++cc) {
            AFrag bf[4];   // B layout: col = l15, dword j -> K = 2j + (j>=4)*8 + hi*8
            #pragma unroll
            for (int i = 0; i < 4; ++i) {
                #pragma unroll
                for (int j = 0; j < 8; ++j)
                    bf[i].u[j] = *(const uint32_t*)
                        &bt[(16 * i + l15) * BT_STRIDE + cc * 32 + 2 * j + ((j >= 4) ? 8 : 0) + hi * 8];
            }
            #pragma unroll
            for (int i = 0; i < 4; ++i)
                acc[i] = __builtin_amdgcn_wmma_f32_16x16x32_f16(false, a[cc].h, false, bf[i].h,
                                                                (short)0, acc[i], false, false);
        }

        // Transpose D through LDS: st[m_local][oc_local], then 16B stores.
        // Same-wave DS ordering makes the store->load->store sequence safe.
        #pragma unroll
        for (int i = 0; i < 4; ++i) {
            #pragma unroll
            for (int r = 0; r < 8; ++r)
                st[wave][l15 * 16 + r + hi * 8] = (_Float16)(acc[i][r] + bv[r]);
            uint4 outv = *(const uint4*)&st[wave][l15 * 16 + hi * 8];
            *(uint4*)&Yout[(size_t)(mt + 16 * i + l15) * OC + ocbase + hi * 8] = outv;
        }
        __syncthreads();   // bt reuse next iteration
    }
}

// ---------------------------------------------------------------------------
// 6) Per-channel sum / sumsq over Y f16 [m][OC], deterministic two-stage
//    reduction (no atomics). Each thread owns a fixed 8-channel group and
//    accumulates in registers over coalesced uint4 rows; block partials go
//    to ps/pq[block][oc] in fixed order.
// ---------------------------------------------------------------------------
__global__ __launch_bounds__(256) void stats_kernel(const _Float16* __restrict__ Y,
                                                    int OC, int rows,
                                                    float* __restrict__ ps,
                                                    float* __restrict__ pq) {
    __shared__ float ls[2048];   // rpb * OC == 2048 always
    __shared__ float lq[2048];

    const int ocg   = OC >> 3;          // uint4 groups per row (16 or 32)
    const int rpb   = 256 / ocg;        // row slots per block (16 or 8)
    const int group = threadIdx.x & (ocg - 1);
    const int rslot = threadIdx.x / ocg;

    float s[8] = {0, 0, 0, 0, 0, 0, 0, 0};
    float q[8] = {0, 0, 0, 0, 0, 0, 0, 0};
    for (int row = blockIdx.x * rpb + rslot; row < rows; row += gridDim.x * rpb) {
        H8 raw; raw.v = *(const uint4*)(Y + (size_t)row * OC + group * 8);
        #pragma unroll
        for (int j = 0; j < 8; ++j) {
            float v = (float)raw.h[j];
            s[j] += v;
            q[j] += v * v;
        }
    }
    #pragma unroll
    for (int j = 0; j < 8; ++j) {
        ls[rslot * OC + group * 8 + j] = s[j];
        lq[rslot * OC + group * 8 + j] = q[j];
    }
    __syncthreads();
    const int oc = threadIdx.x;
    if (oc < OC) {
        float ts = 0.0f, tq = 0.0f;
        for (int r = 0; r < rpb; ++r) {     // fixed order -> deterministic
            ts += ls[r * OC + oc];
            tq += lq[r * OC + oc];
        }
        ps[(size_t)blockIdx.x * 256 + oc] = ts;
        pq[(size_t)blockIdx.x * 256 + oc] = tq;
    }
}

// ---------------------------------------------------------------------------
// 7) BN affine params: reduce per-block partials (fixed order), then
//    scale = g*rsqrt(var+eps), shift = beta - mean*scale.
// ---------------------------------------------------------------------------
__global__ void bn_params_kernel(const float* __restrict__ ps, const float* __restrict__ pq,
                                 int nblocks,
                                 const float* __restrict__ gamma, const float* __restrict__ beta,
                                 int OC, float invM,
                                 float* __restrict__ scale, float* __restrict__ shift) {
    int oc = threadIdx.x;
    if (oc < OC) {
        float s = 0.0f, q = 0.0f;
        for (int bsub = 0; bsub < nblocks; ++bsub) {
            s += ps[(size_t)bsub * 256 + oc];
            q += pq[(size_t)bsub * 256 + oc];
        }
        float m   = s * invM;
        float var = q * invM - m * m;
        float sf  = gamma[oc] * rsqrtf(var + 1e-5f);
        scale[oc] = sf;
        shift[oc] = beta[oc] - m * sf;
    }
}

// ---------------------------------------------------------------------------
// 8) Final: norm+ReLU on Y3 then max over K; write new_points [B][G][256].
// ---------------------------------------------------------------------------
__global__ __launch_bounds__(256) void final_kernel(const _Float16* __restrict__ Y3,
                                                    const float* __restrict__ scale,
                                                    const float* __restrict__ shift,
                                                    float* __restrict__ out_np) {
    const int bg = blockIdx.x;
    const int b  = bg >> 9;
    const int g  = bg & (Gc - 1);
    const int oc = threadIdx.x;
    const float s = scale[oc], t = shift[oc];
    float m = 0.0f;                        // ReLU output >= 0
    for (int k = 0; k < Kc; ++k) {
        size_t mrow = ((size_t)(b * Kc + k) * Gc + g);
        float v = (float)Y3[mrow * 256 + oc];
        v = s * v + t;
        v = v > 0.0f ? v : 0.0f;
        if (v > m) m = v;
    }
    out_np[(size_t)bg * 256 + oc] = m;
}

// ---------------------------------------------------------------------------
// Launcher
// ---------------------------------------------------------------------------
extern "C" void kernel_launch(void* const* d_in, const int* in_sizes, int n_in,
                              void* d_out, int out_size, void* d_ws, size_t ws_size,
                              hipStream_t stream) {
    (void)in_sizes; (void)n_in; (void)out_size; (void)ws_size;

    const float* xyz    = (const float*)d_in[0];
    const float* points = (const float*)d_in[1];
    const float* W1  = (const float*)d_in[2];
    const float* b1  = (const float*)d_in[3];
    const float* g1  = (const float*)d_in[4];
    const float* be1 = (const float*)d_in[5];
    const float* W2  = (const float*)d_in[6];
    const float* b2  = (const float*)d_in[7];
    const float* g2  = (const float*)d_in[8];
    const float* be2 = (const float*)d_in[9];
    const float* W3  = (const float*)d_in[10];
    const float* b3  = (const float*)d_in[11];
    const float* g3  = (const float*)d_in[12];
    const float* be3 = (const float*)d_in[13];

    float* out     = (float*)d_out;
    float* centers = out;                              // [B*G*3]
    float* out_np  = out + (size_t)Bc * Gc * 3;        // [B*G*256]

    char* p = (char*)d_ws;
    auto alloc = [&](size_t bytes) -> char* {
        char* r = p;
        p += (bytes + 255) & ~(size_t)255;
        return r;
    };
    int*      knn_idx = (int*)     alloc(sizeof(int)      * (size_t)Bc * Gc * Kc);
    _Float16* X0      = (_Float16*)alloc(sizeof(_Float16) * (size_t)MTOT * 128);
    _Float16* Y1      = (_Float16*)alloc(sizeof(_Float16) * (size_t)MTOT * 128);
    _Float16* Y2      = (_Float16*)alloc(sizeof(_Float16) * (size_t)MTOT * 128);
    _Float16* Y3      = (_Float16*)alloc(sizeof(_Float16) * (size_t)MTOT * 256);
    _Float16* W16a    = (_Float16*)alloc(sizeof(_Float16) * 128 * 128);
    _Float16* W16b    = (_Float16*)alloc(sizeof(_Float16) * 128 * 128);
    _Float16* W16c    = (_Float16*)alloc(sizeof(_Float16) * 256 * 128);
    float*    ps      = (float*)   alloc(sizeof(float) * 1024 * 256);  // partial sums
    float*    pq      = (float*)   alloc(sizeof(float) * 1024 * 256);  // partial sumsq
    float*    prm     = (float*)   alloc(sizeof(float) * 6 * 256);     // scale/shift x3

    const float invM = 1.0f / (float)MTOT;

    fps_kernel   <<<Bc,        1024, 0, stream>>>(xyz, centers);
    knn_kernel   <<<Bc * Gc,    256, 0, stream>>>(xyz, centers, knn_idx);
    gather_kernel<<<MTOT / 2,   256, 0, stream>>>(points, knn_idx, X0);

    cvt_f16_kernel<<<(128 * 128 + 255) / 256, 256, 0, stream>>>(W1, W16a, 128 * 128);
    cvt_f16_kernel<<<(128 * 128 + 255) / 256, 256, 0, stream>>>(W2, W16b, 128 * 128);
    cvt_f16_kernel<<<(256 * 128 + 255) / 256, 256, 0, stream>>>(W3, W16c, 256 * 128);

    // Layer 1 (raw input, no norm on load)
    gemm_kernel<<<dim3(MTOT / (16 * GEMM_MTILES), 1), 256, 0, stream>>>(
        X0, W16a, b1, prm, prm, 0, 128, Y1);
    stats_kernel<<<1024, 256, 0, stream>>>(Y1, 128, MTOT, ps, pq);
    bn_params_kernel<<<1, 256, 0, stream>>>(ps, pq, 1024, g1, be1, 128, invM,
                                            prm + 0, prm + 256);
    // Layer 2 (fused BN1+ReLU on load)
    gemm_kernel<<<dim3(MTOT / (16 * GEMM_MTILES), 1), 256, 0, stream>>>(
        Y1, W16b, b2, prm + 0, prm + 256, 1, 128, Y2);
    stats_kernel<<<1024, 256, 0, stream>>>(Y2, 128, MTOT, ps, pq);
    bn_params_kernel<<<1, 256, 0, stream>>>(ps, pq, 1024, g2, be2, 128, invM,
                                            prm + 512, prm + 768);
    // Layer 3 (OC=256 -> grid.y=2, fused BN2+ReLU on load)
    gemm_kernel<<<dim3(MTOT / (16 * GEMM_MTILES), 2), 256, 0, stream>>>(
        Y2, W16c, b3, prm + 512, prm + 768, 1, 256, Y3);
    stats_kernel<<<1024, 256, 0, stream>>>(Y3, 256, MTOT, ps, pq);
    bn_params_kernel<<<1, 256, 0, stream>>>(ps, pq, 1024, g3, be3, 256, invM,
                                            prm + 1024, prm + 1280);
    // BN3 + ReLU + max over K
    final_kernel<<<Bc * Gc, 256, 0, stream>>>(Y3, prm + 1024, prm + 1280, out_np);
}
